// PSI_Full_35622458753025
// MI455X (gfx1250) — compile-verified, matrix operand order checked by reading, not run on hardware
//
#include <hip/hip_runtime.h>
#include <hip/hip_bf16.h>

typedef __attribute__((ext_vector_type(16))) _Float16 v16h;
typedef __attribute__((ext_vector_type(8)))  _Float16 v8h;
typedef __attribute__((ext_vector_type(8)))  float    v8f;
typedef __attribute__((ext_vector_type(4)))  unsigned int u32x4;
typedef __attribute__((ext_vector_type(8)))  int          i32x8;
typedef __attribute__((ext_vector_type(4)))  int          i32x4;

#define BM 128
#define BN 64
#define BK 32
#define LDT 40   // padded LDS row stride in f16 (32 + 8) -> conflict-free b128 reads

// ---------------------------------------------------------------------------
// CDNA5 async global->LDS helpers (ASYNCcnt-tracked, VGPR-bypassing)
// ---------------------------------------------------------------------------
__device__ __forceinline__ uint32_t lds_off(const void* p) {
    // low 32 bits of a generic pointer into the LDS aperture == wave-relative
    // LDS byte offset (aperture base has zero low-32 bits)
    return (uint32_t)(uintptr_t)p;
}

__device__ __forceinline__ void async_ld_b128(uint32_t ldsDst, const void* gsrc) {
    asm volatile("global_load_async_to_lds_b128 %0, %1, off"
                 :: "v"(ldsDst), "v"(gsrc)
                 : "memory");
}

__device__ __forceinline__ void wait_async_2() {
    asm volatile("s_wait_asynccnt 0x2" ::: "memory");
}
__device__ __forceinline__ void wait_async_0() {
    asm volatile("s_wait_asynccnt 0x0" ::: "memory");
}

// ---------------------------------------------------------------------------
// CDNA5 Tensor Data Mover: DMA a 64xBK f16 tile of Wt[N,K] into LDS with
// hardware padding (16 DWORDs data + 4 DWORDs pad = LDT=40 f16 row stride).
// Issued per-wave (EXEC ignored); caller must restrict to one wave.
// ---------------------------------------------------------------------------
__device__ __forceinline__ void tdm_load_b_tile(uint32_t ldsDst, const _Float16* gsrc, int K) {
    const unsigned long long ga = (unsigned long long)(uintptr_t)gsrc;
    u32x4 g0;
    g0[0] = 1u;                                        // count=1 valid descriptor
    g0[1] = ldsDst;                                    // lds_addr (bytes)
    g0[2] = (unsigned int)ga;                          // global_addr[31:0]
    g0[3] = (unsigned int)(ga >> 32) | 0x80000000u;    // global_addr[56:32] | type=2<<30
    i32x8 g1;
    g1[0] = (int)((1u << 16)      // data_size = 1 -> 2 bytes
                | (1u << 20)      // pad_enable
                | (3u << 22)      // pad_interval: 16 DWORDs (64B) between pads
                | (3u << 25));    // pad_amount:   4 DWORDs (16B) of pad
    g1[1] = (int)(32u << 16);     // tensor_dim0[15:0] = 32 (K elems per row)
    g1[2] = (int)(64u << 16);     // tensor_dim0[31:16]=0 | tensor_dim1[15:0] = 64 rows
    g1[3] = (int)(32u << 16);     // tensor_dim1[31:16]=0 | tile_dim0 = 32
    g1[4] = 64;                   // tile_dim1 = 64 | tile_dim2 = 0
    g1[5] = K;                    // tensor_dim0_stride[31:0] = K elements
    g1[6] = 0;                    // stride[47:32]=0 | tensor_dim1_stride[15:0]=0
    g1[7] = 0;
    i32x4 z4 = {0, 0, 0, 0};      // groups 2/3 unused (2D tensor)
    i32x8 z8 = {0, 0, 0, 0, 0, 0, 0, 0};
    __builtin_amdgcn_tensor_load_to_lds(g0, g1, z4, z4, z8, 0);
}

// ---------------------------------------------------------------------------
// helpers
// ---------------------------------------------------------------------------
__device__ __forceinline__ float apply_epi(float v, int epi) {
    switch (epi) {
        case 1: return 5.0f / (1.0f + __expf(-v));                         // sigmoid * 5
        case 2: return 1.0f / (1.0f + __expf(-v));                         // sigmoid
        case 3: return 0.5f * v * (1.0f + erff(v * 0.70710678118654752f)); // exact gelu
        default: return v;
    }
}

// Load a 16x32 f16 fragment (A-layout per CDNA5 ISA) for the 16-row group.
// lanes 0-15: K=0..7 (lo) and 16..23 (hi); lanes 16-31: K=8..15 and 24..31.
__device__ __forceinline__ v16h ld_frag(const _Float16* rowBase, int ll, int hi, int ld) {
    const _Float16* p = rowBase + ll * ld + hi * 8;
    v8h lo = *(const v8h*)(p);
    v8h hh = *(const v8h*)(p + 16);
    v16h r;
#pragma unroll
    for (int i = 0; i < 8; ++i) { r[i] = lo[i]; r[i + 8] = hh[i]; }
    return r;
}

// ---------------------------------------------------------------------------
// f32 -> f16 convert
// ---------------------------------------------------------------------------
__global__ void cvt_f16(const float* __restrict__ in, _Float16* __restrict__ out, size_t n) {
    size_t i = (size_t)blockIdx.x * blockDim.x + threadIdx.x;
    if (i < n) out[i] = (_Float16)in[i];
}

// ---------------------------------------------------------------------------
// W[K,N] f32  ->  Wt[N,K] f16   (tiled LDS transpose, 256 threads as 32x8)
// ---------------------------------------------------------------------------
__global__ __launch_bounds__(256) void transpose_to_f16(const float* __restrict__ W,
                                                        _Float16* __restrict__ Wt,
                                                        int K, int N) {
    __shared__ float t[32][33];
    const int tx = threadIdx.x & 31;
    const int ty = threadIdx.x >> 5; // 0..7
    const int nb = blockIdx.x * 32;
    const int kb = blockIdx.y * 32;
#pragma unroll
    for (int j = 0; j < 32; j += 8)
        t[ty + j][tx] = W[(size_t)(kb + ty + j) * N + nb + tx];
    __syncthreads();
#pragma unroll
    for (int j = 0; j < 32; j += 8)
        Wt[(size_t)(nb + ty + j) * K + kb + tx] = (_Float16)t[tx][ty + j];
}

// ---------------------------------------------------------------------------
// C[M,N] = epi(A[M,K]_f16 @ Wt[N,K]_f16 + bias) (+ resid) -> f32 or f16
// block: 256 threads (8 wave32s, 4x2), tile 128x64, K-step 32.
// A tiles: double-buffered async global->LDS copies (ASYNCcnt).
// B tiles: Tensor Data Mover DMA with hardware LDS padding (TENSORcnt).
// ---------------------------------------------------------------------------
__global__ __launch_bounds__(256) void wmma_gemm_f16(
    const _Float16* __restrict__ A, const _Float16* __restrict__ Wt,
    const float* __restrict__ bias, float* __restrict__ outF,
    _Float16* __restrict__ outH, const float* __restrict__ resid,
    int M, int N, int K, int epi) {
    __shared__ _Float16 As[2][BM * LDT];
    __shared__ _Float16 Bs[2][BN * LDT];

    const int tid  = threadIdx.x;
    const int m0   = blockIdx.y * BM;
    const int n0   = blockIdx.x * BN;
    const int wid  = tid >> 5;
    const int lane = tid & 31;
    const int ll   = lane & 15;
    const int hi   = lane >> 4;
    const int wm   = (wid >> 1) * 32; // wave row offset within block tile
    const int wn   = (wid & 1) * 32;  // wave col offset within block tile

    const int ra  = tid >> 1;        // 0..127 (A rows, one per thread)
    const int seg = (tid & 1) * 16;  // 0,16 (f16 elements; 32B halves of a row)

    // per-thread global A sources: each thread copies two b128 (one row of 32 f16
    // is covered by 2 threads x 16 f16)
    const _Float16* aSrc0 = A + (size_t)(m0 + ra) * K + seg;
    const _Float16* aSrc1 = aSrc0 + 8;                 // second 16B of this half

    uint32_t aDst0[2], aDst1[2], bTile[2];
#pragma unroll
    for (int s = 0; s < 2; ++s) {
        aDst0[s] = lds_off(&As[s][(size_t)ra * LDT + seg]);
        aDst1[s] = aDst0[s] + 16;
        bTile[s] = lds_off(&Bs[s][0]);
    }
    const _Float16* bSrc = Wt + (size_t)n0 * K;        // TDM handles the whole tile

    v8f c00 = {}, c01 = {}, c10 = {}, c11 = {};

    const int nk = K / BK;
    // preload tile 0: 2 async ops (A) + 1 TDM op (B, wave 0 only)
    async_ld_b128(aDst0[0], aSrc0);
    async_ld_b128(aDst1[0], aSrc1);
    if (wid == 0) tdm_load_b_tile(bTile[0], bSrc, K);

    for (int kt = 0; kt < nk; ++kt) {
        const int buf = kt & 1;
        if (kt + 1 < nk) {
            const int k1 = (kt + 1) * BK;
            async_ld_b128(aDst0[buf ^ 1], aSrc0 + k1);
            async_ld_b128(aDst1[buf ^ 1], aSrc1 + k1);
            if (wid == 0) tdm_load_b_tile(bTile[buf ^ 1], bSrc + k1, K);
            wait_async_2();                            // in-order: tile kt's A landed
            __builtin_amdgcn_s_wait_tensorcnt(1);      // in-order: tile kt's B landed
        } else {
            wait_async_0();
            __builtin_amdgcn_s_wait_tensorcnt(0);
        }
        __syncthreads();      // tile kt visible to all waves

        v16h a0 = ld_frag(&As[buf][(size_t)wm * LDT], ll, hi, LDT);
        v16h a1 = ld_frag(&As[buf][(size_t)(wm + 16) * LDT], ll, hi, LDT);
        v16h b0 = ld_frag(&Bs[buf][(size_t)wn * LDT], ll, hi, LDT);
        v16h b1 = ld_frag(&Bs[buf][(size_t)(wn + 16) * LDT], ll, hi, LDT);

        c00 = __builtin_amdgcn_wmma_f32_16x16x32_f16(false, a0, false, b0, (short)0, c00, false, false);
        c01 = __builtin_amdgcn_wmma_f32_16x16x32_f16(false, a0, false, b1, (short)0, c01, false, false);
        c10 = __builtin_amdgcn_wmma_f32_16x16x32_f16(false, a1, false, b0, (short)0, c10, false, false);
        c11 = __builtin_amdgcn_wmma_f32_16x16x32_f16(false, a1, false, b1, (short)0, c11, false, false);

        __syncthreads();      // WAR: all waves done reading buf before it is refilled
    }

    // epilogue: C 16x16 f32 layout — VGPR r: lanes0-15 -> M=r, lanes16-31 -> M=r+8; N=lane%16
    v8f accs[4] = {c00, c01, c10, c11};
#pragma unroll
    for (int mt = 0; mt < 2; ++mt) {
#pragma unroll
        for (int nt = 0; nt < 2; ++nt) {
            v8f acc = accs[mt * 2 + nt];
            const int col = n0 + wn + nt * 16 + ll;
            const float bv = bias ? bias[col] : 0.0f;
#pragma unroll
            for (int r = 0; r < 8; ++r) {
                const int row = m0 + wm + mt * 16 + r + hi * 8;
                float v = apply_epi(acc[r] + bv, epi);
                const size_t idx = (size_t)row * N + col;
                if (resid) v += resid[idx];
                if (outF) outF[idx] = v;
                else      outH[idx] = (_Float16)v;
            }
        }
    }
    (void)M;
}

// ---------------------------------------------------------------------------
// Sequential phase / complex-memory scan along S. One thread per (b,d) channel.
// ---------------------------------------------------------------------------
__global__ __launch_bounds__(256) void psi_scan(
    const float* __restrict__ x, const float* __restrict__ omega,
    const float* __restrict__ gate, const float* __restrict__ mag,
    const float* __restrict__ phi_init, const float* __restrict__ qoff,
    const float* __restrict__ int_scale, float* __restrict__ ctx,
    int S, int D) {
    const int d = blockIdx.x * blockDim.x + threadIdx.x;
    const int b = blockIdx.y;
    const float isc = fabsf(int_scale[d]);

    float accPhi = 0.f, accR = 0.f, accI = 0.f, accM = 0.f;
    const size_t base  = (size_t)b * S * D + d;
    const size_t cbase = (size_t)b * S * (4 * D) + d;

    for (int s = 0; s < S; ++s) {
        const size_t idx = base + (size_t)s * D;
        const float om = omega[idx];
        const float gt = gate[idx];
        const float mg = mag[idx];
        const float xi = x[idx];
        const float pi = phi_init[idx];
        const float qo = qoff[idx];

        accPhi += gt * om * isc;
        const float phi = pi + accPhi;
        float sp, cp;
        __sincosf(phi, &sp, &cp);

        const float wc = mg * xi;
        accR += wc * cp;
        accI += wc * sp;
        accM += mg;
        const float rs = rsqrtf(accM + 1e-8f);
        const float mr = accR * rs;
        const float mi = accI * rs;

        float sq, cq;
        __sincosf(phi + qo, &sq, &cq);
        const float rr = mr * cq + mi * sq;
        const float ri = mi * cq - mr * sq;

        const size_t c0 = cbase + (size_t)s * (4 * D);
        ctx[c0]           = xi * cp;
        ctx[c0 + D]       = xi * sp;
        ctx[c0 + 2 * D]   = rr;
        ctx[c0 + 3 * D]   = ri;
    }
}

// ---------------------------------------------------------------------------
// LayerNorm over F=2048 features, one block per token, output f16
// ---------------------------------------------------------------------------
__global__ __launch_bounds__(256) void layernorm_f16(
    const float* __restrict__ ctx, const float* __restrict__ g,
    const float* __restrict__ bta, _Float16* __restrict__ out, int F) {
    __shared__ float s1[256], s2[256];
    const int t = blockIdx.x;
    const float* row = ctx + (size_t)t * F;
    float v[8];
    float sum = 0.f, sq = 0.f;
#pragma unroll
    for (int j = 0; j < 8; ++j) {
        float xv = row[threadIdx.x + j * 256];
        v[j] = xv; sum += xv; sq += xv * xv;
    }
    s1[threadIdx.x] = sum; s2[threadIdx.x] = sq;
    __syncthreads();
    for (int off = 128; off > 0; off >>= 1) {
        if (threadIdx.x < off) {
            s1[threadIdx.x] += s1[threadIdx.x + off];
            s2[threadIdx.x] += s2[threadIdx.x + off];
        }
        __syncthreads();
    }
    const float mean = s1[0] / F;
    const float var  = s2[0] / F - mean * mean;
    const float rstd = rsqrtf(var + 1e-5f);
    _Float16* orow = out + (size_t)t * F;
#pragma unroll
    for (int j = 0; j < 8; ++j) {
        const int i = threadIdx.x + j * 256;
        orow[i] = (_Float16)((v[j] - mean) * rstd * g[i] + bta[i]);
    }
}

// ---------------------------------------------------------------------------
// host launcher
// ---------------------------------------------------------------------------
extern "C" void kernel_launch(void* const* d_in, const int* in_sizes, int n_in,
                              void* d_out, int out_size, void* d_ws, size_t ws_size,
                              hipStream_t stream) {
    (void)in_sizes; (void)n_in; (void)out_size; (void)ws_size;
    constexpr int Bc = 8, S = 4096, D = 512;
    constexpr size_t T = (size_t)Bc * S; // 32768 tokens

    const float* x        = (const float*)d_in[0];
    const float* W_omega  = (const float*)d_in[1];  const float* b_omega = (const float*)d_in[2];
    const float* W_p1     = (const float*)d_in[3];  const float* b_p1    = (const float*)d_in[4];
    const float* W_p2     = (const float*)d_in[5];  const float* b_p2    = (const float*)d_in[6];
    const float* W_gate   = (const float*)d_in[7];  const float* b_gate  = (const float*)d_in[8];
    const float* int_scale= (const float*)d_in[9];
    const float* W_mag    = (const float*)d_in[10]; const float* b_mag   = (const float*)d_in[11];
    const float* W_qoff   = (const float*)d_in[12]; const float* b_qoff  = (const float*)d_in[13];
    const float* ln_g     = (const float*)d_in[14]; const float* ln_b    = (const float*)d_in[15];
    const float* W_o1     = (const float*)d_in[16]; const float* b_o1    = (const float*)d_in[17];
    const float* W_o2     = (const float*)d_in[18]; const float* b_o2    = (const float*)d_in[19];
    float* out = (float*)d_out;

    char* ws = (char*)d_ws;
    size_t off = 0;
    auto alloc = [&](size_t bytes) -> char* {
        char* p = ws + off;
        off = (off + bytes + 255) & ~(size_t)255;
        return p;
    };

    _Float16* xh     = (_Float16*)alloc(T * D * 2);
    _Float16* wtOm   = (_Float16*)alloc((size_t)D * D * 2);
    _Float16* wtP1   = (_Float16*)alloc((size_t)D * D * 2);
    _Float16* wtP2   = (_Float16*)alloc((size_t)D * D * 2);
    _Float16* wtGate = (_Float16*)alloc((size_t)D * D * 2);
    _Float16* wtMag  = (_Float16*)alloc((size_t)D * D * 2);
    _Float16* wtQoff = (_Float16*)alloc((size_t)D * D * 2);
    _Float16* wtO1   = (_Float16*)alloc((size_t)2048 * 1024 * 2);
    _Float16* wtO2   = (_Float16*)alloc((size_t)1024 * 512 * 2);
    float* omega = (float*)alloc(T * D * 4);
    float* magn  = (float*)alloc(T * D * 4);
    float* gatef = (float*)alloc(T * D * 4);
    float* qofff = (float*)alloc(T * D * 4);
    float* phii  = (float*)alloc(T * D * 4);
    _Float16* p1g = (_Float16*)alloc(T * D * 2);
    float* ctx = (float*)alloc(T * 4 * D * 4);
    // safe reuse: scan/LN inputs are dead by the time these are written
    _Float16* ctxh = (_Float16*)omega; // 128MB fits omega+magn (contiguous 128MB)
    _Float16* h1   = (_Float16*)gatef; // 64MB fits gate

    // --- conversions ---
    {
        size_t n = T * D;
        cvt_f16<<<dim3((unsigned)((n + 255) / 256)), 256, 0, stream>>>(x, xh, n);
    }
    transpose_to_f16<<<dim3(D / 32, D / 32), 256, 0, stream>>>(W_omega, wtOm, D, D);
    transpose_to_f16<<<dim3(D / 32, D / 32), 256, 0, stream>>>(W_p1,    wtP1, D, D);
    transpose_to_f16<<<dim3(D / 32, D / 32), 256, 0, stream>>>(W_p2,    wtP2, D, D);
    transpose_to_f16<<<dim3(D / 32, D / 32), 256, 0, stream>>>(W_gate,  wtGate, D, D);
    transpose_to_f16<<<dim3(D / 32, D / 32), 256, 0, stream>>>(W_mag,   wtMag, D, D);
    transpose_to_f16<<<dim3(D / 32, D / 32), 256, 0, stream>>>(W_qoff,  wtQoff, D, D);
    transpose_to_f16<<<dim3(1024 / 32, 2048 / 32), 256, 0, stream>>>(W_o1, wtO1, 2048, 1024);
    transpose_to_f16<<<dim3(512 / 32, 1024 / 32), 256, 0, stream>>>(W_o2, wtO2, 1024, 512);

    // --- projection GEMMs ---
    dim3 g1(D / BN, (unsigned)(T / BM)); // (8, 256)
    wmma_gemm_f16<<<g1, 256, 0, stream>>>(xh, wtOm,   b_omega, omega, nullptr, nullptr, (int)T, D, D, 0);
    wmma_gemm_f16<<<g1, 256, 0, stream>>>(xh, wtMag,  b_mag,   magn,  nullptr, nullptr, (int)T, D, D, 1);
    wmma_gemm_f16<<<g1, 256, 0, stream>>>(xh, wtGate, b_gate,  gatef, nullptr, nullptr, (int)T, D, D, 2);
    wmma_gemm_f16<<<g1, 256, 0, stream>>>(xh, wtQoff, b_qoff,  qofff, nullptr, nullptr, (int)T, D, D, 0);
    wmma_gemm_f16<<<g1, 256, 0, stream>>>(xh, wtP1,   b_p1,    nullptr, p1g,   nullptr, (int)T, D, D, 3);
    wmma_gemm_f16<<<g1, 256, 0, stream>>>(p1g, wtP2,  b_p2,    phii,  nullptr, nullptr, (int)T, D, D, 0);

    // --- sequential scan over S ---
    psi_scan<<<dim3(D / 256, Bc), 256, 0, stream>>>(x, omega, gatef, magn, phii, qofff,
                                                    int_scale, ctx, S, D);

    // --- layernorm (f16 output) ---
    layernorm_f16<<<dim3((unsigned)T), 256, 0, stream>>>(ctx, ln_g, ln_b, ctxh, 4 * D);

    // --- output MLP ---
    wmma_gemm_f16<<<dim3(1024 / BN, (unsigned)(T / BM)), 256, 0, stream>>>(
        ctxh, wtO1, b_o1, nullptr, h1, nullptr, (int)T, 1024, 2048, 3);
    wmma_gemm_f16<<<dim3(512 / BN, (unsigned)(T / BM)), 256, 0, stream>>>(
        h1, wtO2, b_o2, out, nullptr, x, (int)T, 512, 1024, 0);
}
